// Attention_14654428414644
// MI455X (gfx1250) — matrix operand order repaired and checked by reading, not compiled, not run on hardware
//
#include <hip/hip_runtime.h>

#define SQ   4096
#define EDIM 1024
#define DDIM 64
#define BATCH 4

typedef __attribute__((ext_vector_type(16))) __bf16 v16bf;
typedef __attribute__((ext_vector_type(8)))  float  v8f;

__device__ __forceinline__ v8f wmma_bf16(v16bf a, v16bf b, v8f c) {
  return __builtin_amdgcn_wmma_f32_16x16x32_bf16(
      /*neg_a=*/false, a, /*neg_b=*/false, b,
      /*c_mod=*/(short)0, c, /*reuse_a=*/false, /*reuse_b=*/false);
}

// B fragment: 16 contiguous bf16 at p (K = base..base+15 for this half-wave)
__device__ __forceinline__ v16bf load_b_frag(const __bf16* p) {
  union { v16bf v; uint4 q[2]; } u;
  u.q[0] = *reinterpret_cast<const uint4*>(p);
  u.q[1] = *reinterpret_cast<const uint4*>(p + 8);
  return u.v;
}

// A fragment: 8 contiguous bf16 at p+off1 (K base..base+7) and p+off2 (K base+8..+15)
__device__ __forceinline__ v16bf load_a_frag(const __bf16* p, int off1, int off2) {
  union { v16bf v; uint4 q[2]; } u;
  u.q[0] = *reinterpret_cast<const uint4*>(p + off1);
  u.q[1] = *reinterpret_cast<const uint4*>(p + off2);
  return u.v;
}

// ---------------- Phase 0: fp32 -> bf16 bulk conversion ----------------------
__global__ void __launch_bounds__(256) cvt_kernel(const float* __restrict__ src,
                                                  __bf16* __restrict__ dst,
                                                  int n /* multiple of 8 */) {
  const int i = (blockIdx.x * 256 + threadIdx.x) * 8;
  if (i >= n) return;
  const float4 a = *reinterpret_cast<const float4*>(src + i);
  const float4 b = *reinterpret_cast<const float4*>(src + i + 4);
  union { __bf16 h[8]; uint4 q; } u;
  u.h[0] = (__bf16)a.x; u.h[1] = (__bf16)a.y;
  u.h[2] = (__bf16)a.z; u.h[3] = (__bf16)a.w;
  u.h[4] = (__bf16)b.x; u.h[5] = (__bf16)b.y;
  u.h[6] = (__bf16)b.z; u.h[7] = (__bf16)b.w;
  *reinterpret_cast<uint4*>(dst + i) = u.q;
}

// ---------------- Phase 1: Q/K/V projections (bf16 in/out, Q pre-scaled) ----
// Software pipeline unrolled x2: buffers ping-pong by name, so loop-carried
// fragments are defined directly by loads (no v_mov rotation), and WMMAs
// overlap with the next step's global_load_b128 clause.
__global__ void __launch_bounds__(128) proj_kernel(
    const __bf16* __restrict__ xbf,
    const __bf16* __restrict__ wqbf, const __bf16* __restrict__ wkbf,
    const __bf16* __restrict__ wvbf,
    __bf16* __restrict__ qbf, __bf16* __restrict__ kbf,
    __bf16* __restrict__ vbf) {
  const int lane = threadIdx.x & 31;
  const int wave = threadIdx.x >> 5;
  const int lm = lane & 15;
  const int hf = lane >> 4;
  const int m0 = (blockIdx.x * 4 + wave) * 16;
  const int mat = blockIdx.y;
  const __bf16* W = (mat == 0) ? wqbf : (mat == 1) ? wkbf : wvbf;
  __bf16* out = (mat == 0) ? qbf : (mat == 1) ? kbf : vbf;
  const float scale = (mat == 0) ? 0.125f : 1.0f;   // 1/sqrt(64) folded into Q

  v8f acc[4];
#pragma unroll
  for (int t = 0; t < 4; ++t) acc[t] = {};

  const __bf16* xrow = xbf + (long)(m0 + lm) * EDIM;
  const __bf16* wrow[4];
#pragma unroll
  for (int t = 0; t < 4; ++t)
    wrow[t] = W + (long)(t * 16 + lm) * EDIM + hf * 16;

  // prologue: buffer 0 holds fragments for e0 = 0
  v16bf a0 = load_a_frag(xrow, hf * 16, hf * 16 + 8);
  v16bf b0[4];
#pragma unroll
  for (int t = 0; t < 4; ++t) b0[t] = load_b_frag(wrow[t]);

  for (int e0 = 0; e0 < EDIM - 64; e0 += 64) {
    // load step e0+32 into buffer 1, compute step e0 from buffer 0
    const v16bf a1 = load_a_frag(xrow, e0 + 32 + hf * 16, e0 + 32 + hf * 16 + 8);
    v16bf b1[4];
#pragma unroll
    for (int t = 0; t < 4; ++t) b1[t] = load_b_frag(wrow[t] + e0 + 32);
#pragma unroll
    for (int t = 0; t < 4; ++t) acc[t] = wmma_bf16(a0, b0[t], acc[t]);

    // load step e0+64 into buffer 0, compute step e0+32 from buffer 1
    a0 = load_a_frag(xrow, e0 + 64 + hf * 16, e0 + 64 + hf * 16 + 8);
#pragma unroll
    for (int t = 0; t < 4; ++t) b0[t] = load_b_frag(wrow[t] + e0 + 64);
#pragma unroll
    for (int t = 0; t < 4; ++t) acc[t] = wmma_bf16(a1, b1[t], acc[t]);
  }

  // tail: buffer 0 holds step EDIM-64; load and compute final step EDIM-32
  {
    const v16bf a1 = load_a_frag(xrow, EDIM - 32 + hf * 16, EDIM - 32 + hf * 16 + 8);
    v16bf b1[4];
#pragma unroll
    for (int t = 0; t < 4; ++t) b1[t] = load_b_frag(wrow[t] + EDIM - 32);
#pragma unroll
    for (int t = 0; t < 4; ++t) acc[t] = wmma_bf16(a0, b0[t], acc[t]);
#pragma unroll
    for (int t = 0; t < 4; ++t) acc[t] = wmma_bf16(a1, b1[t], acc[t]);
  }

#pragma unroll
  for (int t = 0; t < 4; ++t) {
#pragma unroll
    for (int j = 0; j < 8; ++j) {
      const int row = m0 + j + hf * 8;
      const int col = t * 16 + lm;
      out[(long)row * DDIM + col] = (__bf16)(acc[t][j] * scale);
    }
  }
}

// ---------------- Phase 2: flash attention ----------------------------------
__global__ void __launch_bounds__(128) attn_kernel(
    const __bf16* __restrict__ qbf, const __bf16* __restrict__ kbf,
    const __bf16* __restrict__ vbf, const int* __restrict__ mask,
    float* __restrict__ out) {
  __shared__ __align__(16) __bf16 Ks[64 * 64];    // [key_local][d]
  __shared__ __align__(16) __bf16 Vt[64 * 64];    // [d][key_local]
  __shared__ __align__(16) __bf16 Ps[4][16 * 64]; // per-wave P staging

  const int lane = threadIdx.x & 31;
  const int wave = threadIdx.x >> 5;
  const int lm = lane & 15;
  const int hf = lane >> 4;
  const int b = blockIdx.y;
  const int q0 = (blockIdx.x * 4 + wave) * 16;

  // Q fragments (stay resident; Q is pre-scaled)
  const __bf16* qrow = qbf + (long)(b * SQ + q0 + lm) * DDIM;
  v16bf qa[2];
#pragma unroll
  for (int c = 0; c < 2; ++c)
    qa[c] = load_a_frag(qrow, 32 * c + hf * 16, 32 * c + hf * 16 + 8);

  v8f o[4];
  float mrun[8], lrun[8];
#pragma unroll
  for (int t = 0; t < 4; ++t) o[t] = {};
#pragma unroll
  for (int j = 0; j < 8; ++j) { mrun[j] = -1e30f; lrun[j] = 0.0f; }

  const __bf16* kbase = kbf + (long)b * SQ * DDIM;
  const __bf16* vbase = vbf + (long)b * SQ * DDIM;
  const int* mbase = mask + (long)b * SQ * SQ;

  for (int kb = 0; kb < SQ; kb += 64) {
    __syncthreads();
    {
      // K tile: row-major, vectorized b128 copies
      const int r = threadIdx.x >> 1;          // key row 0..63
      const int ch = (threadIdx.x & 1) * 32;   // half-row
      const uint4* ksrc = reinterpret_cast<const uint4*>(kbase + (long)(kb + r) * DDIM + ch);
      uint4* kdst = reinterpret_cast<uint4*>(&Ks[r * 64 + ch]);
#pragma unroll
      for (int i = 0; i < 4; ++i) kdst[i] = ksrc[i];

      // V tile: transposed in registers, written as b64 (4 keys per store)
      const int kq = (threadIdx.x & 15) * 4;   // key offset 0..60
      const int dg = (threadIdx.x >> 4) * 8;   // d offset 0..56
      union { uint4 q; __bf16 h[8]; } vr[4];
#pragma unroll
      for (int kk = 0; kk < 4; ++kk)
        vr[kk].q = *reinterpret_cast<const uint4*>(vbase + (long)(kb + kq + kk) * DDIM + dg);
#pragma unroll
      for (int i = 0; i < 8; ++i) {
        union { __bf16 h[4]; unsigned long long u; } pk;
        pk.h[0] = vr[0].h[i]; pk.h[1] = vr[1].h[i];
        pk.h[2] = vr[2].h[i]; pk.h[3] = vr[3].h[i];
        *reinterpret_cast<unsigned long long*>(&Vt[(dg + i) * 64 + kq]) = pk.u;
      }
      if (kb + 64 < SQ) {
        __builtin_prefetch(kbase + (long)(kb + 64 + r) * DDIM + ch, 0, 0);
        __builtin_prefetch(vbase + (long)(kb + 64 + kq) * DDIM + dg, 0, 0);
      }
    }
    __syncthreads();

    // S = Q K^T   (16 queries x 64 keys), f32 accum
    v8f sc[4];
#pragma unroll
    for (int t = 0; t < 4; ++t) {
      v8f z = {};
      const v16bf b0 = load_b_frag(&Ks[(t * 16 + lm) * 64 + hf * 16]);
      z = wmma_bf16(qa[0], b0, z);
      const v16bf b1 = load_b_frag(&Ks[(t * 16 + lm) * 64 + 32 + hf * 16]);
      sc[t] = wmma_bf16(qa[1], b1, z);
    }

    // mask + online softmax (a score row lives in 16 lanes of one half-wave)
#pragma unroll
    for (int j = 0; j < 8; ++j) {
      const int row = q0 + j + hf * 8;
      const int* mrow = mbase + (long)row * SQ + kb;
      float tmax = -3.0e38f;
#pragma unroll
      for (int t = 0; t < 4; ++t) {
        float s = sc[t][j];
        const int mv = mrow[t * 16 + lm];
        s = (mv != 0) ? s : -3.0e38f;
        sc[t][j] = s;
        tmax = fmaxf(tmax, s);
      }
#pragma unroll
      for (int off = 8; off >= 1; off >>= 1)
        tmax = fmaxf(tmax, __shfl_xor(tmax, off, 32));
      const float mnew = fmaxf(mrun[j], tmax);
      const float alpha = __expf(mrun[j] - mnew);
      mrun[j] = mnew;
      float psum = 0.0f;
#pragma unroll
      for (int t = 0; t < 4; ++t) {
        const float p = __expf(sc[t][j] - mnew);
        sc[t][j] = p;
        psum += p;
      }
#pragma unroll
      for (int off = 8; off >= 1; off >>= 1)
        psum += __shfl_xor(psum, off, 32);
      lrun[j] = lrun[j] * alpha + psum;
#pragma unroll
      for (int t = 0; t < 4; ++t) o[t][j] *= alpha;
    }

    // P -> bf16 -> LDS, reload in A-matrix layout
    __bf16* Pw = &Ps[wave][0];
#pragma unroll
    for (int t = 0; t < 4; ++t) {
#pragma unroll
      for (int j = 0; j < 8; ++j)
        Pw[(j + hf * 8) * 64 + t * 16 + lm] = (__bf16)sc[t][j];
    }
    asm volatile("s_wait_dscnt 0" ::: "memory");

    v16bf pa[2];
#pragma unroll
    for (int c = 0; c < 2; ++c)
      pa[c] = load_a_frag(Pw + lm * 64, 32 * c + hf * 16, 32 * c + hf * 16 + 8);

    // O += P V  (V staged transposed, so B fragments are contiguous in LDS)
#pragma unroll
    for (int t2 = 0; t2 < 4; ++t2) {
      const v16bf bv0 = load_b_frag(&Vt[(t2 * 16 + lm) * 64 + hf * 16]);
      o[t2] = wmma_bf16(pa[0], bv0, o[t2]);
      const v16bf bv1 = load_b_frag(&Vt[(t2 * 16 + lm) * 64 + 32 + hf * 16]);
      o[t2] = wmma_bf16(pa[1], bv1, o[t2]);
    }
  }

  // normalize and store fp32 output
#pragma unroll
  for (int j = 0; j < 8; ++j) {
    const float inv = 1.0f / lrun[j];
#pragma unroll
    for (int t2 = 0; t2 < 4; ++t2) {
      const int row = q0 + j + hf * 8;
      out[(long)(b * SQ + row) * DDIM + t2 * 16 + lm] = o[t2][j] * inv;
    }
  }
}

extern "C" void kernel_launch(void* const* d_in, const int* in_sizes, int n_in,
                              void* d_out, int out_size, void* d_ws, size_t ws_size,
                              hipStream_t stream) {
  const float* x  = (const float*)d_in[0];
  const float* Wq = (const float*)d_in[1];
  const float* Wk = (const float*)d_in[2];
  const float* Wv = (const float*)d_in[3];
  const int* mask = (const int*)d_in[4];
  float* out = (float*)d_out;

  const long R  = (long)BATCH * SQ;    // 16384 rows
  const long NX = R * EDIM;            // 16,777,216 x elements
  const long NW = (long)DDIM * EDIM;   // 65,536 weight elements

  __bf16* qbf  = (__bf16*)d_ws;        // R*64
  __bf16* kbf  = qbf + R * DDIM;
  __bf16* vbf  = kbf + R * DDIM;
  __bf16* xbf  = vbf + R * DDIM;       // R*EDIM (32 MB)
  __bf16* wqbf = xbf + NX;
  __bf16* wkbf = wqbf + NW;
  __bf16* wvbf = wkbf + NW;

  // Phase 0: bulk fp32 -> bf16 conversion (hoists all cvt out of GEMM loops)
  cvt_kernel<<<(unsigned)(NX / 8 / 256), 256, 0, stream>>>(x,  xbf,  (int)NX);
  cvt_kernel<<<(unsigned)(NW / 8 / 256), 256, 0, stream>>>(Wq, wqbf, (int)NW);
  cvt_kernel<<<(unsigned)(NW / 8 / 256), 256, 0, stream>>>(Wk, wkbf, (int)NW);
  cvt_kernel<<<(unsigned)(NW / 8 / 256), 256, 0, stream>>>(Wv, wvbf, (int)NW);

  // Phase 1: projections (ping-pong pipelined load_b128 + v_wmma inner loop)
  dim3 g1((unsigned)(R / 64), 3);
  proj_kernel<<<g1, 128, 0, stream>>>(xbf, wqbf, wkbf, wvbf, qbf, kbf, vbf);

  // Phase 2: flash attention
  dim3 g2(SQ / 64, BATCH);
  attn_kernel<<<g2, 128, 0, stream>>>(qbf, kbf, vbf, mask, out);
}
